// EditDistNeuralModel_48369921688175
// MI455X (gfx1250) — compile-verified, matrix operand order checked by reading, not run on hardware
//
#include <hip/hip_runtime.h>
#include <math.h>

// ---------------------------------------------------------------------------
// EditDistNeuralModel for MI455X (gfx1250, wave32, WMMA)
// Outputs (flat, in order): scores[256*256*961], exp_ec[256*256*961],
//                           entropy[1], alpha_last[1]
// ---------------------------------------------------------------------------

#define LL 256          // sequence length (T == V == 256)
#define DD 64           // model dim
#define CC 961          // class count (1 + 30 + 30 + 900)
#define CP 976          // padded to 61 * 16 for WMMA N-tiling
#define TVC ((size_t)LL * LL * CC)
#define NEGINF (-INFINITY)

typedef __attribute__((ext_vector_type(16))) _Float16 v16h;
typedef __attribute__((ext_vector_type(8)))  _Float16 v8h;
typedef __attribute__((ext_vector_type(8)))  float    v8f;
typedef __attribute__((ext_vector_type(4)))  float    v4f;   // ext-vector (NT-store capable)

__device__ __forceinline__ float lae(float a, float b) {
  float m = fmaxf(a, b);
  if (m == NEGINF) return NEGINF;
  return m + __logf(__expf(a - m) + __expf(b - m));
}

// ----------------------------- encoder kernels -----------------------------

__global__ void k_embed(const int* __restrict__ tok, const float* __restrict__ emb,
                        float* __restrict__ x) {
  int idx = blockIdx.x * blockDim.x + threadIdx.x;
  if (idx >= LL * DD) return;
  int l = idx / DD, d = idx % DD;
  float div = __expf(-logf(10000.0f) * (float)(d & ~1) / (float)DD);
  float ang = (float)l * div;
  float pe = (d & 1) ? __cosf(ang) : __sinf(ang);
  x[idx] = emb[tok[l] * DD + d] * 8.0f + pe;   // sqrt(64) == 8
}

// out[m][n] = act( A[m,:K] . W[n,:K] + bias[n] )
__global__ void k_gemm(const float* __restrict__ A, int lda,
                       const float* __restrict__ W, int ldw,
                       const float* __restrict__ bias, float* __restrict__ out,
                       int M, int N, int K, int do_relu) {
  int idx = blockIdx.x * blockDim.x + threadIdx.x;
  if (idx >= M * N) return;
  int m = idx / N, n = idx % N;
  const float* a = A + (size_t)m * lda;
  const float* w = W + (size_t)n * ldw;
  float acc = bias ? bias[n] : 0.0f;
  for (int k = 0; k < K; ++k) acc += a[k] * w[k];
  if (do_relu) acc = fmaxf(acc, 0.0f);
  out[(size_t)m * N + n] = acc;
}

// self-attention: one block per (query l, head h); 256 threads (one per key)
__global__ void k_attn(const float* __restrict__ qkv, float* __restrict__ ao) {
  int l = blockIdx.x, h = blockIdx.y, j = threadIdx.x;
  __shared__ float att[256];
  __shared__ float red[256];
  const float* q = qkv + (size_t)l * 192 + h * 16;
  const float* k = qkv + (size_t)j * 192 + 64 + h * 16;
  float s = 0.f;
  #pragma unroll
  for (int d = 0; d < 16; ++d) s += q[d] * k[d];
  s *= 0.25f;                       // 1/sqrt(16)
  red[j] = s;
  __syncthreads();
  for (int off = 128; off > 0; off >>= 1) {
    if (j < off) red[j] = fmaxf(red[j], red[j + off]);
    __syncthreads();
  }
  float mx = red[0];
  __syncthreads();
  float p = __expf(s - mx);
  att[j] = p; red[j] = p;
  __syncthreads();
  for (int off = 128; off > 0; off >>= 1) {
    if (j < off) red[j] += red[j + off];
    __syncthreads();
  }
  float inv = 1.0f / red[0];
  __syncthreads();
  if (j < 16) {
    float acc = 0.f;
    for (int kk = 0; kk < 256; ++kk)
      acc += att[kk] * qkv[(size_t)kk * 192 + 128 + h * 16 + j];
    ao[(size_t)l * DD + h * 16 + j] = acc * inv;
  }
}

// layernorm(xa + xb) * s + b ; one block (64 threads) per row
__global__ void k_ln(const float* __restrict__ xa, const float* __restrict__ xb,
                     const float* __restrict__ s, const float* __restrict__ b,
                     float* __restrict__ out) {
  int l = blockIdx.x, d = threadIdx.x;
  __shared__ float red[64];
  float x = xa[(size_t)l * DD + d] + xb[(size_t)l * DD + d];
  red[d] = x; __syncthreads();
  for (int off = 32; off > 0; off >>= 1) { if (d < off) red[d] += red[d + off]; __syncthreads(); }
  float mean = red[0] * (1.0f / DD);
  __syncthreads();
  float c = x - mean;
  red[d] = c * c; __syncthreads();
  for (int off = 32; off > 0; off >>= 1) { if (d < off) red[d] += red[d + off]; __syncthreads(); }
  float var = red[0] * (1.0f / DD);
  out[(size_t)l * DD + d] = c * rsqrtf(var + 1e-5f) * s[d] + b[d];
}

// ----------------------------- scorer kernels ------------------------------

// convert W2 (961 x 64 f32) -> padded (976 x 64 f16)
__global__ void k_w2h(const float* __restrict__ w2, _Float16* __restrict__ w2h) {
  int idx = blockIdx.x * blockDim.x + threadIdx.x;
  if (idx >= CP * DD) return;
  int c = idx / DD;
  w2h[idx] = (c < CC) ? (_Float16)w2[idx] : (_Float16)0.0f;
}

// online-softmax merge of (mx,sum) pairs; -inf safe
__device__ __forceinline__ void osm_merge(float& mx, float& sum, float omx, float osum) {
  float nm = fmaxf(mx, omx);
  float s1 = (mx  == NEGINF) ? 0.f : sum  * __expf(mx  - nm);
  float s2 = (omx == NEGINF) ? 0.f : osum * __expf(omx - nm);
  mx = nm; sum = s1 + s2;
}

// Fused scorer: logits(t,v,c) = W2[c,:] . relu(HA[t]+HB[v]+b1) + b2[c],
// log-softmax over c, entropy partials and the 3 sparse gathers.
// Phase 1: WMMA tiles -> per-row online (max,sumexp) stats only.
// Phase 2: recompute WMMA tiles (8 GFLOP is free vs 252 MB of traffic),
//          write scores = logit - lse with NONTEMPORAL stores (never re-read).
// Block = 256 threads (8 waves); tile = fixed t, 64 consecutive v, all 961 c.
// Wave w: M-subtile mi = w&3 (16 rows), N-tiles nj with parity (w>>2).
__global__ void __launch_bounds__(256)
k_score_wmma(const float* __restrict__ HA, const float* __restrict__ HB,
             const float* __restrict__ b1, const _Float16* __restrict__ w2h,
             const float* __restrict__ b2, const int* __restrict__ ar,
             const int* __restrict__ en, float* __restrict__ scores,
             float* __restrict__ a_del, float* __restrict__ a_ins,
             float* __restrict__ a_sub, float* __restrict__ ent1) {
  int t  = blockIdx.x >> 2;
  int v0 = (blockIdx.x & 3) * 64;
  __shared__ _Float16 h16[64][64];     // 8 KB: A-operand tile in f16
  __shared__ float s_mx[8][16];        // per-wave per-row stats
  __shared__ float s_sum[8][16];
  __shared__ float s_lse[64];
  __shared__ int   s_eid[64];          // en[v0+m]
  __shared__ float red[256];
  int tid = threadIdx.x;
  {
    int m  = tid >> 2;                 // v-row within tile
    int k0 = (tid & 3) * 16;
    const float* ha = HA + (size_t)t * DD;
    const float* hb = HB + (size_t)(v0 + m) * DD;
    for (int k = k0; k < k0 + 16; ++k)
      h16[m][k] = (_Float16)fmaxf(ha[k] + hb[k] + b1[k], 0.0f);
  }
  if (tid < 64) s_eid[tid] = en[v0 + tid];
  __syncthreads();

  int w = tid >> 5, lane = tid & 31;
  int mi = w & 3;                      // M-subtile (fixed per wave)
  int njp = w >> 2;                    // N-tile parity
  int nlo = lane & 15;                 // row (A) / class-in-tile (B)
  int hi  = lane >> 4;                 // ISA: lanes 16-31 hold K = 8..15, 24..31
  int arid = ar[t];
  int did  = 1 + arid;
  int sub0 = 61 + 30 * arid;

  // A operands (loop-invariant per wave), per ISA 16-bit A layout
  v16h a, a_hi;
  {
    v8h a0 = *(const v8h*)&h16[mi * 16 + nlo][hi * 8];
    v8h a1 = *(const v8h*)&h16[mi * 16 + nlo][hi * 8 + 16];
    v8h a2 = *(const v8h*)&h16[mi * 16 + nlo][32 + hi * 8];
    v8h a3 = *(const v8h*)&h16[mi * 16 + nlo][32 + hi * 8 + 16];
    #pragma unroll
    for (int e = 0; e < 8; ++e) {
      a[e] = a0[e];    a[e + 8] = a1[e];
      a_hi[e] = a2[e]; a_hi[e + 8] = a3[e];
    }
  }

  // ---------------- phase 1: per-row online softmax stats ----------------
  float mx8[8], sum8[8];
  #pragma unroll
  for (int i = 0; i < 8; ++i) { mx8[i] = NEGINF; sum8[i] = 0.f; }

  for (int nj = njp; nj < 61; nj += 2) {
    int c = nj * 16 + nlo;
    v8f acc = {};
    const _Float16* wrow = w2h + (size_t)c * DD + hi * 8;
    v8h bb0 = *(const v8h*)wrow;
    v8h bb1 = *(const v8h*)(wrow + 16);
    v8h bb2 = *(const v8h*)(wrow + 32);
    v8h bb3 = *(const v8h*)(wrow + 48);
    v16h b, b_hi;
    #pragma unroll
    for (int e = 0; e < 8; ++e) {
      b[e] = bb0[e];    b[e + 8] = bb1[e];
      b_hi[e] = bb2[e]; b_hi[e + 8] = bb3[e];
    }
    acc = __builtin_amdgcn_wmma_f32_16x16x32_f16(false, a, false, b, (short)0, acc, false, false);
    acc = __builtin_amdgcn_wmma_f32_16x16x32_f16(false, a_hi, false, b_hi, (short)0, acc, false, false);
    if (c < CC) {
      float bc = b2[c];
      #pragma unroll
      for (int i = 0; i < 8; ++i) {
        float val = acc[i] + bc;
        float nm = fmaxf(mx8[i], val);
        sum8[i] = ((mx8[i] == NEGINF) ? 0.f : sum8[i] * __expf(mx8[i] - nm)) + __expf(val - nm);
        mx8[i] = nm;
      }
    }
  }
  // cross-lane merge within each 16-lane half (rows differ between halves)
  #pragma unroll
  for (int m = 1; m < 16; m <<= 1) {
    #pragma unroll
    for (int i = 0; i < 8; ++i) {
      float omx  = __shfl_xor(mx8[i],  m, 32);
      float osum = __shfl_xor(sum8[i], m, 32);
      osm_merge(mx8[i], sum8[i], omx, osum);
    }
  }
  if (nlo == 0) {
    #pragma unroll
    for (int i = 0; i < 8; ++i) {
      s_mx[w][i + hi * 8]  = mx8[i];
      s_sum[w][i + hi * 8] = sum8[i];
    }
  }
  __syncthreads();
  if (tid < 64) {                       // combine the wave pair sharing mi
    int wmi = tid >> 4, r = tid & 15;
    float mx = s_mx[wmi][r], sm = s_sum[wmi][r];
    osm_merge(mx, sm, s_mx[wmi + 4][r], s_sum[wmi + 4][r]);
    s_lse[tid] = mx + __logf(sm);
  }
  __syncthreads();

  // ---------------- phase 2: recompute, normalize, emit ----------------
  float ent = 0.f;
  for (int nj = njp; nj < 61; nj += 2) {
    int c = nj * 16 + nlo;
    v8f acc = {};
    const _Float16* wrow = w2h + (size_t)c * DD + hi * 8;
    v8h bb0 = *(const v8h*)wrow;
    v8h bb1 = *(const v8h*)(wrow + 16);
    v8h bb2 = *(const v8h*)(wrow + 32);
    v8h bb3 = *(const v8h*)(wrow + 48);
    v16h b, b_hi;
    #pragma unroll
    for (int e = 0; e < 8; ++e) {
      b[e] = bb0[e];    b[e + 8] = bb1[e];
      b_hi[e] = bb2[e]; b_hi[e + 8] = bb3[e];
    }
    acc = __builtin_amdgcn_wmma_f32_16x16x32_f16(false, a, false, b, (short)0, acc, false, false);
    acc = __builtin_amdgcn_wmma_f32_16x16x32_f16(false, a_hi, false, b_hi, (short)0, acc, false, false);
    if (c < CC) {
      float bc = b2[c];
      size_t base = ((size_t)t * LL) * CC + (size_t)c;
      #pragma unroll
      for (int i = 0; i < 8; ++i) {     // C/D layout: VGPR i -> row i (+8 if hi)
        int lrow = mi * 16 + i + hi * 8;
        int vg = v0 + lrow;
        float s = acc[i] + bc - s_lse[lrow];
        __builtin_nontemporal_store(s, &scores[base + (size_t)vg * CC]);
        ent += s * __expf(s);
        int eid = s_eid[lrow];
        int tv = t * LL + vg;
        if (c == did)        a_del[tv] = s;
        if (c == 31 + eid)   a_ins[tv] = s;
        if (c == sub0 + eid) a_sub[tv] = s;
      }
    }
  }
  // deterministic block entropy partial
  red[tid] = ent; __syncthreads();
  for (int off = 128; off > 0; off >>= 1) { if (tid < off) red[tid] += red[tid + off]; __syncthreads(); }
  if (tid == 0) ent1[blockIdx.x] = -red[0];
}

// deterministic 256-way tree sum
__global__ void k_reduce(const float* __restrict__ in, float* __restrict__ out, int n) {
  __shared__ float red[256];
  int tid = threadIdx.x;
  int base = blockIdx.x * 256;
  red[tid] = (base + tid < n) ? in[base + tid] : 0.f;
  __syncthreads();
  for (int off = 128; off > 0; off >>= 1) { if (tid < off) red[tid] += red[tid + off]; __syncthreads(); }
  if (tid == 0) out[blockIdx.x] = red[0];
}

// alpha DP via anti-diagonal wavefront; single 256-thread block, rolling LDS
__global__ void __launch_bounds__(256)
k_alpha(const float* __restrict__ ai, const float* __restrict__ ad,
        const float* __restrict__ asu, float* __restrict__ alpha) {
  __shared__ float buf[3][256];
  int t = threadIdx.x;
  for (int d = 0; d < 2 * LL - 1; ++d) {
    int j = d - t;
    if (j >= 0 && j < LL) {
      float val;
      if (t == 0 && j == 0) {
        val = 0.f;
      } else if (t == 0) {
        val = buf[(d + 2) % 3][0] + ai[j];                  // cumsum of ins row
      } else if (j == 0) {
        val = buf[(d + 2) % 3][t - 1] + ad[(size_t)t * LL]; // del column
      } else {
        float left = buf[(d + 2) % 3][t];        // alpha[t][j-1]
        float up   = buf[(d + 2) % 3][t - 1];    // alpha[t-1][j]
        float diag = buf[(d + 1) % 3][t - 1];    // alpha[t-1][j-1]
        int idx = t * LL + j;
        val = lae(lae(ai[idx] + left, ad[idx] + up), asu[idx] + diag);
      }
      buf[d % 3][t] = val;
      alpha[t * LL + j] = val;
    }
    __syncthreads();
  }
}

// zero-fill exp(ec) region (nontemporal ext-vector stores; write-once)
__global__ void k_fill0(v4f* __restrict__ p, size_t n4) {
  size_t i = (size_t)blockIdx.x * blockDim.x + threadIdx.x;
  if (i < n4) {
    v4f z = {0.f, 0.f, 0.f, 0.f};
    __builtin_nontemporal_store(z, &p[i]);
  }
}

// expected counts: beta cancels in the per-(t,v) normalization, so
// exp(ec) is a 3-way softmax over {alpha[t-1,v]+s_del, alpha[t,v-1]+s_ins,
// alpha[t-1,v-1]+s_sub}; scatter up to 3 values per (t,v).
__global__ void k_scatter(const float* __restrict__ alpha, const float* __restrict__ a_del,
                          const float* __restrict__ a_ins, const float* __restrict__ a_sub,
                          const int* __restrict__ ar, const int* __restrict__ en,
                          float* __restrict__ ec, float* __restrict__ extra) {
  int tv = blockIdx.x * blockDim.x + threadIdx.x;
  if (tv >= LL * LL) return;
  int t = tv >> 8, v = tv & 255;
  float dv = (t >= 1) ? alpha[(t - 1) * LL + v] + a_del[tv] : NEGINF;
  float iv = (v >= 1) ? alpha[t * LL + (v - 1)] + a_ins[tv] : NEGINF;
  float sv = (t >= 1 && v >= 1) ? alpha[(t - 1) * LL + (v - 1)] + a_sub[tv] : NEGINF;
  float lse = lae(lae(dv, iv), sv);
  if (lse != NEGINF) {
    float* base = ec + (size_t)tv * CC;
    if (t >= 1)            base[1 + ar[t]]                = __expf(dv - lse);
    if (v >= 1)            base[31 + en[v]]               = __expf(iv - lse);
    if (t >= 1 && v >= 1)  base[61 + 30 * ar[t] + en[v]]  = __expf(sv - lse);
  }
  if (tv == LL * LL - 1) extra[1] = alpha[LL * LL - 1];  // alpha[-1,-1]
}

// --------------------------- workspace layout (floats) ---------------------

constexpr size_t OFF_X     = 0;
constexpr size_t OFF_QKV   = OFF_X + 256 * 64;
constexpr size_t OFF_AO    = OFF_QKV + 256 * 192;
constexpr size_t OFF_O2    = OFF_AO + 256 * 64;
constexpr size_t OFF_X1    = OFF_O2 + 256 * 64;
constexpr size_t OFF_FF    = OFF_X1 + 256 * 64;
constexpr size_t OFF_ARV   = OFF_FF + 256 * 128;
constexpr size_t OFF_ENV   = OFF_ARV + 256 * 64;
constexpr size_t OFF_HA    = OFF_ENV + 256 * 64;
constexpr size_t OFF_HB    = OFF_HA + 256 * 64;
constexpr size_t OFF_ADEL  = OFF_HB + 256 * 64;
constexpr size_t OFF_AINS  = OFF_ADEL + 65536;
constexpr size_t OFF_ASUB  = OFF_AINS + 65536;
constexpr size_t OFF_ALPHA = OFF_ASUB + 65536;
constexpr size_t OFF_ENT1  = OFF_ALPHA + 65536;   // 1024 block partials
constexpr size_t OFF_ENT2  = OFF_ENT1 + 1024;     // 4 second-level partials
constexpr size_t OFF_W2H   = OFF_ENT2 + 256;      // reinterpreted as _Float16*

static void run_encoder(const float* emb, const float* ipw, const float* ipb,
                        const float* ow, const float* ob,
                        const float* l1s, const float* l1b,
                        const float* l2s, const float* l2b,
                        const float* f1w, const float* f1b,
                        const float* f2w, const float* f2b,
                        const int* tok, float* ws, float* outv, hipStream_t stream) {
  float* x   = ws + OFF_X;
  float* qkv = ws + OFF_QKV;
  float* ao  = ws + OFF_AO;
  float* o2  = ws + OFF_O2;
  float* x1  = ws + OFF_X1;
  float* ff  = ws + OFF_FF;
  k_embed<<<(LL * DD + 255) / 256, 256, 0, stream>>>(tok, emb, x);
  k_gemm<<<(256 * 192 + 255) / 256, 256, 0, stream>>>(x, 64, ipw, 64, ipb, qkv, 256, 192, 64, 0);
  k_attn<<<dim3(256, 4), 256, 0, stream>>>(qkv, ao);
  k_gemm<<<(256 * 64 + 255) / 256, 256, 0, stream>>>(ao, 64, ow, 64, ob, o2, 256, 64, 64, 0);
  k_ln<<<256, 64, 0, stream>>>(x, o2, l1s, l1b, x1);
  k_gemm<<<(256 * 128 + 255) / 256, 256, 0, stream>>>(x1, 64, f1w, 64, f1b, ff, 256, 128, 64, 1);
  k_gemm<<<(256 * 64 + 255) / 256, 256, 0, stream>>>(ff, 128, f2w, 128, f2b, o2, 256, 64, 128, 0);
  k_ln<<<256, 64, 0, stream>>>(x1, o2, l2s, l2b, outv);
}

extern "C" void kernel_launch(void* const* d_in, const int* in_sizes, int n_in,
                              void* d_out, int out_size, void* d_ws, size_t ws_size,
                              hipStream_t stream) {
  // setup_inputs() flattening order:
  // 0: ar_sent, 1: en_sent, 2..14: ar_enc{emb,in_proj_w,in_proj_b,out_w,out_b,
  // ln1_s,ln1_b,ln2_s,ln2_b,ff1_w,ff1_b,ff2_w,ff2_b}, 15..27: en_enc (same),
  // 28: w1, 29: b1, 30: w2, 31: b2
  const int* ar = (const int*)d_in[0];
  const int* en = (const int*)d_in[1];
  const float* F[32];
  for (int i = 2; i < 32; ++i) F[i] = (const float*)d_in[i];

  float* ws = (float*)d_ws;
  _Float16* w2h = (_Float16*)(ws + OFF_W2H);
  float* scores = (float*)d_out;
  float* ec     = scores + TVC;
  float* extra  = scores + 2 * TVC;   // [0]=entropy, [1]=alpha[-1,-1]

  // encoders
  run_encoder(F[2], F[3], F[4], F[5], F[6], F[7], F[8], F[9], F[10], F[11],
              F[12], F[13], F[14], ar, ws, ws + OFF_ARV, stream);
  run_encoder(F[15], F[16], F[17], F[18], F[19], F[20], F[21], F[22], F[23],
              F[24], F[25], F[26], F[27], en, ws, ws + OFF_ENV, stream);

  // GEMM1 factorized: HA = arv @ W1[:, :64]^T ; HB = env @ W1[:, 64:]^T
  const float* w1 = F[28];
  k_gemm<<<(256 * 64 + 255) / 256, 256, 0, stream>>>(ws + OFF_ARV, 64, w1, 128,
                                                     nullptr, ws + OFF_HA, 256, 64, 64, 0);
  k_gemm<<<(256 * 64 + 255) / 256, 256, 0, stream>>>(ws + OFF_ENV, 64, w1 + 64, 128,
                                                     nullptr, ws + OFF_HB, 256, 64, 64, 0);

  // W2 -> f16 (padded), then the fused WMMA GEMM + log-softmax + gathers
  k_w2h<<<(CP * DD + 255) / 256, 256, 0, stream>>>(F[30], w2h);
  k_score_wmma<<<LL * 4, 256, 0, stream>>>(ws + OFF_HA, ws + OFF_HB, F[29],
                                           w2h, F[31], ar, en, scores,
                                           ws + OFF_ADEL, ws + OFF_AINS,
                                           ws + OFF_ASUB, ws + OFF_ENT1);

  // alpha wavefront DP (args: ins, del, sub)
  k_alpha<<<1, 256, 0, stream>>>(ws + OFF_AINS, ws + OFF_ADEL, ws + OFF_ASUB,
                                 ws + OFF_ALPHA);

  // deterministic entropy reduction -> extra[0]
  k_reduce<<<4, 256, 0, stream>>>(ws + OFF_ENT1, ws + OFF_ENT2, 1024);
  k_reduce<<<1, 256, 0, stream>>>(ws + OFF_ENT2, extra, 4);

  // exp(ec): nontemporal zero-fill then sparse scatter (+ alpha[-1,-1])
  k_fill0<<<(unsigned)(TVC / 4 / 256), 256, 0, stream>>>((v4f*)ec, TVC / 4);
  k_scatter<<<(LL * LL + 255) / 256, 256, 0, stream>>>(ws + OFF_ALPHA, ws + OFF_ADEL,
                                                       ws + OFF_AINS, ws + OFF_ASUB,
                                                       ar, en, ec, extra);
}
